// DeepseekMoE_16587163697456
// MI455X (gfx1250) — compile-verified, matrix operand order checked by reading, not run on hardware
//
#include <hip/hip_runtime.h>
#include <hip/hip_bf16.h>

// ---------------- constants (match reference shapes) ----------------
constexpr int cG    = 64;     // HQQ quant group size
constexpr int cE    = 16;     // experts
constexpr int cTOPK = 6;
constexpr int cH    = 2048;
constexpr int cI    = 1408;
constexpr int cSH   = 2816;
constexpr int cT    = 4096;   // B*S tokens
constexpr int TILE_M = 128;   // block M-tile (8 waves x 16)
constexpr int TILE_N = 64;
constexpr int KCH    = 64;    // k-chunk == quant group
constexpr int LPAD   = 72;    // LDS row stride (bf16) to dodge bank aliasing
constexpr int NTHREADS = 256; // 8 wave32

typedef __bf16 bf16;
typedef __attribute__((ext_vector_type(16))) __bf16 bf16x16;
typedef __attribute__((ext_vector_type(8)))  __bf16 bf16x8;
typedef __attribute__((ext_vector_type(8)))  float  f32x8;

__device__ __forceinline__ bf16 f2bf(float f) {
  unsigned u = __builtin_bit_cast(unsigned, f);
  unsigned r = u + 0x7FFFu + ((u >> 16) & 1u);        // RNE
  unsigned short h = (unsigned short)(r >> 16);
  return __builtin_bit_cast(bf16, h);
}

// Load one 16x32-bf16 A/B fragment slice for this lane.
// row points at the (M or N) row start; kbase already includes the +8 offset
// for lanes 16..31.  Halves [0..7]=K kbase..kbase+7, [8..15]=K kbase+16..+23.
__device__ __forceinline__ bf16x16 load_frag(const bf16* row, int kbase) {
  bf16x8 lo = *(const bf16x8*)(row + kbase);
  bf16x8 hi = *(const bf16x8*)(row + kbase + 16);
  struct P { bf16x8 a, b; } p{lo, hi};
  return __builtin_bit_cast(bf16x16, p);
}

__device__ __forceinline__ f32x8 wmma_bf16(bf16x16 a, bf16x16 b, f32x8 c) {
  return __builtin_amdgcn_wmma_f32_16x16x32_bf16(false, a, false, b,
                                                 (short)0, c, false, false);
}

__device__ __forceinline__ float silu(float g) {
  return g / (1.f + __expf(-g));
}

// ---------------- tiny setup kernels ----------------
__global__ void zero_counts_kernel(int* counts) {
  if (threadIdx.x < cE) counts[threadIdx.x] = 0;
}

__global__ void prefix_counts_kernel(const int* counts, int* offsets) {
  if (threadIdx.x == 0) {
    int acc = 0;
    for (int e = 0; e < cE; ++e) { offsets[e] = acc; acc += counts[e]; }
  }
}

__global__ __launch_bounds__(256)
void cvt_x_kernel(const float* __restrict__ x, bf16* __restrict__ xb, int n) {
  int i = blockIdx.x * blockDim.x + threadIdx.x;
  if (i < n) xb[i] = f2bf(x[i]);
}

// ---------------- gating: softmax -> top6 -> renorm -> bucket append ---------
__global__ __launch_bounds__(128)
void gate_kernel(const float* __restrict__ x, const float* __restrict__ gw,
                 int* __restrict__ counts, int* __restrict__ tid_list,
                 float* __restrict__ w_list) {
  const int wv = threadIdx.x >> 5;
  const int lane = threadIdx.x & 31;
  const int t = blockIdx.x * 4 + wv;
  if (t >= cT) return;
  const float* xr = x + (size_t)t * cH;

  float logit = -INFINITY;
  for (int e = 0; e < cE; ++e) {
    const float* g = gw + (size_t)e * cH;
    float p = 0.f;
    for (int h = lane; h < cH; h += 32) p += xr[h] * g[h];
    for (int o = 16; o; o >>= 1) p += __shfl_xor(p, o, 32);
    if (lane == e) logit = p;
  }
  float m = logit;
  for (int o = 16; o; o >>= 1) m = fmaxf(m, __shfl_xor(m, o, 32));
  float pe = __expf(logit - m);
  float s = pe;
  for (int o = 16; o; o >>= 1) s += __shfl_xor(s, o, 32);
  float score = pe / s;        // lanes>=16: exp(-inf)=0

  float topv[cTOPK]; int tope[cTOPK]; float wsum = 0.f;
  float sc = score;
  for (int k = 0; k < cTOPK; ++k) {
    float v = sc; int id = lane;
    for (int o = 16; o; o >>= 1) {
      float ov = __shfl_xor(v, o, 32);
      int   oi = __shfl_xor(id, o, 32);
      if (ov > v || (ov == v && oi < id)) { v = ov; id = oi; }
    }
    topv[k] = v; tope[k] = id; wsum += v;
    if (lane == id) sc = -1.f;
  }
  if (lane == 0) {
    float inv = 1.f / (wsum + 1e-20f);
    for (int k = 0; k < cTOPK; ++k) {
      int e = tope[k];
      int slot = atomicAdd(&counts[e], 1);
      tid_list[e * cT + slot] = t;
      w_list[e * cT + slot] = topv[k] * inv;
    }
  }
}

// ---------------- routed gate+up (pipelined dequant, SiLU, weight fold) ------
__global__ __launch_bounds__(NTHREADS)
void moe_gateup_kernel(const bf16* __restrict__ xb,
                       const int* __restrict__ Wq_g, const float* __restrict__ sc_g,
                       const float* __restrict__ zp_g,
                       const int* __restrict__ Wq_u, const float* __restrict__ sc_u,
                       const float* __restrict__ zp_u,
                       const int* __restrict__ counts, const int* __restrict__ offsets,
                       const int* __restrict__ tid_list, const float* __restrict__ w_list,
                       bf16* __restrict__ Y) {
  const int e = blockIdx.z, mt = blockIdx.x, nb = blockIdx.y;
  const int count = counts[e];
  const int m0 = mt * TILE_M;
  if (m0 >= count) return;
  const int off = offsets[e];
  const int tid = threadIdx.x, lane = tid & 31, wv = tid >> 5;

  __shared__ bf16 sBg[TILE_N][LPAD];
  __shared__ bf16 sBu[TILE_N][LPAD];
  __shared__ float sW[TILE_M];

  if (tid < TILE_M) {
    int idx = m0 + tid;
    sW[tid] = (idx < count) ? w_list[e * cT + idx] : 0.f;
  }
  const int idxA = m0 + wv * 16 + (lane & 15);
  const int tA = (idxA < count) ? tid_list[e * cT + idxA] : 0;
  const bf16* xrow = xb + (size_t)tA * cH;

  f32x8 accG[4] = {}; f32x8 accU[4] = {};
  const int nbase = nb * TILE_N;
  const int sn = tid >> 2;              // 0..63 staged row
  const int sk = (tid & 3) * 16;        // 16 codes per thread per matrix
  const int khalf = (lane >> 4) * 8;

  const int* gbase = Wq_g + ((size_t)e * cI + nbase + sn) * cH + sk;
  const int* ubase = Wq_u + ((size_t)e * cI + nbase + sn) * cH + sk;
  const size_t qbase = ((size_t)e * cI + nbase + sn) * (cH / cG);
  constexpr int NKG = cH / KCH;

  int4 cg[4], cu[4];
  float sgv, zsg, suv, zsu;
  {
    const int4* g4 = (const int4*)gbase;
    const int4* u4 = (const int4*)ubase;
#pragma unroll
    for (int j = 0; j < 4; ++j) { cg[j] = g4[j]; cu[j] = u4[j]; }
    sgv = sc_g[qbase]; zsg = zp_g[qbase] * sgv;
    suv = sc_u[qbase]; zsu = zp_u[qbase] * suv;
  }

  for (int kg = 0; kg < NKG; ++kg) {
    // convert+store current chunk
#pragma unroll
    for (int j = 0; j < 4; ++j) {
      sBg[sn][sk + 4 * j + 0] = f2bf((float)cg[j].x * sgv - zsg);
      sBg[sn][sk + 4 * j + 1] = f2bf((float)cg[j].y * sgv - zsg);
      sBg[sn][sk + 4 * j + 2] = f2bf((float)cg[j].z * sgv - zsg);
      sBg[sn][sk + 4 * j + 3] = f2bf((float)cg[j].w * sgv - zsg);
      sBu[sn][sk + 4 * j + 0] = f2bf((float)cu[j].x * suv - zsu);
      sBu[sn][sk + 4 * j + 1] = f2bf((float)cu[j].y * suv - zsu);
      sBu[sn][sk + 4 * j + 2] = f2bf((float)cu[j].z * suv - zsu);
      sBu[sn][sk + 4 * j + 3] = f2bf((float)cu[j].w * suv - zsu);
    }
    __syncthreads();
    // prefetch next chunk (in flight under the WMMAs)
    if (kg + 1 < NKG) {
      const int4* g4 = (const int4*)(gbase + (kg + 1) * KCH);
      const int4* u4 = (const int4*)(ubase + (kg + 1) * KCH);
#pragma unroll
      for (int j = 0; j < 4; ++j) { cg[j] = g4[j]; cu[j] = u4[j]; }
      sgv = sc_g[qbase + kg + 1]; zsg = zp_g[qbase + kg + 1] * sgv;
      suv = sc_u[qbase + kg + 1]; zsu = zp_u[qbase + kg + 1] * suv;
    }
    const int k0 = kg * KCH;
#pragma unroll
    for (int ks = 0; ks < KCH; ks += 32) {
      bf16x16 a = load_frag(xrow, k0 + ks + khalf);
#pragma unroll
      for (int j = 0; j < 4; ++j) {
        bf16x16 bg = load_frag(&sBg[j * 16 + (lane & 15)][0], ks + khalf);
        bf16x16 bu = load_frag(&sBu[j * 16 + (lane & 15)][0], ks + khalf);
        accG[j] = wmma_bf16(a, bg, accG[j]);
        accU[j] = wmma_bf16(a, bu, accU[j]);
      }
    }
    __syncthreads();
  }
  const int colb = lane & 15;
  const int rhi = (lane >> 4) * 8;
#pragma unroll
  for (int j = 0; j < 4; ++j) {
#pragma unroll
    for (int r = 0; r < 8; ++r) {
      int rl = wv * 16 + rhi + r;
      int idx = m0 + rl;
      if (idx < count) {
        float y = silu(accG[j][r]) * accU[j][r] * sW[rl];
        Y[(size_t)(off + idx) * cI + (nbase + j * 16 + colb)] = f2bf(y);
      }
    }
  }
}

// ---------------- routed down-proj (pipelined dequant, atomic scatter) -------
__global__ __launch_bounds__(NTHREADS)
void moe_down_kernel(const bf16* __restrict__ Y,
                     const int* __restrict__ Wq_d, const float* __restrict__ sc_d,
                     const float* __restrict__ zp_d,
                     const int* __restrict__ counts, const int* __restrict__ offsets,
                     const int* __restrict__ tid_list,
                     float* __restrict__ out) {
  const int e = blockIdx.z, mt = blockIdx.x, nb = blockIdx.y;
  const int count = counts[e];
  const int m0 = mt * TILE_M;
  if (m0 >= count) return;
  const int off = offsets[e];
  const int tid = threadIdx.x, lane = tid & 31, wv = tid >> 5;

  __shared__ bf16 sB[TILE_N][LPAD];
  __shared__ int sT[TILE_M];

  if (tid < TILE_M) {
    int idx = m0 + tid;
    sT[tid] = (idx < count) ? tid_list[e * cT + idx] : 0;
  }
  int idxA = m0 + wv * 16 + (lane & 15);
  if (idxA >= count) idxA = count - 1;       // clamp; masked at store
  const bf16* yrow = Y + (size_t)(off + idxA) * cI;

  f32x8 acc[4] = {};
  const int nbase = nb * TILE_N;
  const int sn = tid >> 2, sk = (tid & 3) * 16;
  const int khalf = (lane >> 4) * 8;

  const int* dbase = Wq_d + ((size_t)e * cH + nbase + sn) * cI + sk;
  const size_t qbase = ((size_t)e * cH + nbase + sn) * (cI / cG);
  constexpr int NKG = cI / KCH;

  int4 cd[4]; float sdv, zsd;
  {
    const int4* d4 = (const int4*)dbase;
#pragma unroll
    for (int j = 0; j < 4; ++j) cd[j] = d4[j];
    sdv = sc_d[qbase]; zsd = zp_d[qbase] * sdv;
  }

  for (int kg = 0; kg < NKG; ++kg) {
#pragma unroll
    for (int j = 0; j < 4; ++j) {
      sB[sn][sk + 4 * j + 0] = f2bf((float)cd[j].x * sdv - zsd);
      sB[sn][sk + 4 * j + 1] = f2bf((float)cd[j].y * sdv - zsd);
      sB[sn][sk + 4 * j + 2] = f2bf((float)cd[j].z * sdv - zsd);
      sB[sn][sk + 4 * j + 3] = f2bf((float)cd[j].w * sdv - zsd);
    }
    __syncthreads();
    if (kg + 1 < NKG) {
      const int4* d4 = (const int4*)(dbase + (kg + 1) * KCH);
#pragma unroll
      for (int j = 0; j < 4; ++j) cd[j] = d4[j];
      sdv = sc_d[qbase + kg + 1]; zsd = zp_d[qbase + kg + 1] * sdv;
    }
    const int k0 = kg * KCH;
#pragma unroll
    for (int ks = 0; ks < KCH; ks += 32) {
      bf16x16 a = load_frag(yrow, k0 + ks + khalf);
#pragma unroll
      for (int j = 0; j < 4; ++j) {
        bf16x16 b = load_frag(&sB[j * 16 + (lane & 15)][0], ks + khalf);
        acc[j] = wmma_bf16(a, b, acc[j]);
      }
    }
    __syncthreads();
  }
  const int colb = lane & 15;
  const int rhi = (lane >> 4) * 8;
#pragma unroll
  for (int j = 0; j < 4; ++j) {
#pragma unroll
    for (int r = 0; r < 8; ++r) {
      int rl = wv * 16 + rhi + r;
      int idx = m0 + rl;
      if (idx < count) {
        int t = sT[rl];
        float* dst = &out[(size_t)t * cH + (nbase + j * 16 + colb)];
        __hip_atomic_fetch_add(dst, acc[j][r], __ATOMIC_RELAXED,
                               __HIP_MEMORY_SCOPE_AGENT);
      }
    }
  }
}

// ---------------- shared expert: gate/up (fp32 weights -> bf16 LDS) ----------
__global__ __launch_bounds__(NTHREADS)
void shared_mlp1_kernel(const bf16* __restrict__ xb,
                        const float* __restrict__ Wg, const float* __restrict__ Wu,
                        bf16* __restrict__ S) {
  const int mt = blockIdx.x, nb = blockIdx.y;
  const int m0 = mt * TILE_M, nbase = nb * TILE_N;
  const int tid = threadIdx.x, lane = tid & 31, wv = tid >> 5;

  __shared__ bf16 sBg[TILE_N][LPAD];
  __shared__ bf16 sBu[TILE_N][LPAD];

  const int rowA = m0 + wv * 16 + (lane & 15);
  const bf16* xrow = xb + (size_t)rowA * cH;

  f32x8 accG[4] = {}; f32x8 accU[4] = {};
  const int sn = tid >> 2, sk = (tid & 3) * 16;
  const int khalf = (lane >> 4) * 8;

  const float* gbase = Wg + (size_t)(nbase + sn) * cH + sk;
  const float* ubase = Wu + (size_t)(nbase + sn) * cH + sk;
  constexpr int NKG = cH / KCH;

  float4 cg[4], cu[4];
  {
    const float4* g4 = (const float4*)gbase;
    const float4* u4 = (const float4*)ubase;
#pragma unroll
    for (int j = 0; j < 4; ++j) { cg[j] = g4[j]; cu[j] = u4[j]; }
  }

  for (int kg = 0; kg < NKG; ++kg) {
#pragma unroll
    for (int j = 0; j < 4; ++j) {
      sBg[sn][sk + 4 * j + 0] = f2bf(cg[j].x); sBg[sn][sk + 4 * j + 1] = f2bf(cg[j].y);
      sBg[sn][sk + 4 * j + 2] = f2bf(cg[j].z); sBg[sn][sk + 4 * j + 3] = f2bf(cg[j].w);
      sBu[sn][sk + 4 * j + 0] = f2bf(cu[j].x); sBu[sn][sk + 4 * j + 1] = f2bf(cu[j].y);
      sBu[sn][sk + 4 * j + 2] = f2bf(cu[j].z); sBu[sn][sk + 4 * j + 3] = f2bf(cu[j].w);
    }
    __syncthreads();
    if (kg + 1 < NKG) {
      const float4* g4 = (const float4*)(gbase + (kg + 1) * KCH);
      const float4* u4 = (const float4*)(ubase + (kg + 1) * KCH);
#pragma unroll
      for (int j = 0; j < 4; ++j) { cg[j] = g4[j]; cu[j] = u4[j]; }
    }
    const int k0 = kg * KCH;
#pragma unroll
    for (int ks = 0; ks < KCH; ks += 32) {
      bf16x16 a = load_frag(xrow, k0 + ks + khalf);
#pragma unroll
      for (int j = 0; j < 4; ++j) {
        bf16x16 bg = load_frag(&sBg[j * 16 + (lane & 15)][0], ks + khalf);
        bf16x16 bu = load_frag(&sBu[j * 16 + (lane & 15)][0], ks + khalf);
        accG[j] = wmma_bf16(a, bg, accG[j]);
        accU[j] = wmma_bf16(a, bu, accU[j]);
      }
    }
    __syncthreads();
  }
  const int colb = lane & 15;
  const int rhi = (lane >> 4) * 8;
#pragma unroll
  for (int j = 0; j < 4; ++j) {
#pragma unroll
    for (int r = 0; r < 8; ++r) {
      int rl = wv * 16 + rhi + r;
      float y = silu(accG[j][r]) * accU[j][r];
      S[(size_t)(m0 + rl) * cSH + (nbase + j * 16 + colb)] = f2bf(y);
    }
  }
}

// ---------------- shared expert: down (plain stores establish d_out) ---------
__global__ __launch_bounds__(NTHREADS)
void shared_down_kernel(const bf16* __restrict__ S,
                        const float* __restrict__ Wd,
                        float* __restrict__ out) {
  const int mt = blockIdx.x, nb = blockIdx.y;
  const int m0 = mt * TILE_M, nbase = nb * TILE_N;
  const int tid = threadIdx.x, lane = tid & 31, wv = tid >> 5;

  __shared__ bf16 sB[TILE_N][LPAD];

  const int rowA = m0 + wv * 16 + (lane & 15);
  const bf16* srow = S + (size_t)rowA * cSH;

  f32x8 acc[4] = {};
  const int sn = tid >> 2, sk = (tid & 3) * 16;
  const int khalf = (lane >> 4) * 8;

  const float* dbase = Wd + (size_t)(nbase + sn) * cSH + sk;
  constexpr int NKG = cSH / KCH;

  float4 cd[4];
  {
    const float4* d4 = (const float4*)dbase;
#pragma unroll
    for (int j = 0; j < 4; ++j) cd[j] = d4[j];
  }

  for (int kg = 0; kg < NKG; ++kg) {
#pragma unroll
    for (int j = 0; j < 4; ++j) {
      sB[sn][sk + 4 * j + 0] = f2bf(cd[j].x); sB[sn][sk + 4 * j + 1] = f2bf(cd[j].y);
      sB[sn][sk + 4 * j + 2] = f2bf(cd[j].z); sB[sn][sk + 4 * j + 3] = f2bf(cd[j].w);
    }
    __syncthreads();
    if (kg + 1 < NKG) {
      const float4* d4 = (const float4*)(dbase + (kg + 1) * KCH);
#pragma unroll
      for (int j = 0; j < 4; ++j) cd[j] = d4[j];
    }
    const int k0 = kg * KCH;
#pragma unroll
    for (int ks = 0; ks < KCH; ks += 32) {
      bf16x16 a = load_frag(srow, k0 + ks + khalf);
#pragma unroll
      for (int j = 0; j < 4; ++j) {
        bf16x16 b = load_frag(&sB[j * 16 + (lane & 15)][0], ks + khalf);
        acc[j] = wmma_bf16(a, b, acc[j]);
      }
    }
    __syncthreads();
  }
  const int colb = lane & 15;
  const int rhi = (lane >> 4) * 8;
#pragma unroll
  for (int j = 0; j < 4; ++j) {
#pragma unroll
    for (int r = 0; r < 8; ++r) {
      int rl = wv * 16 + rhi + r;
      out[(size_t)(m0 + rl) * cH + (nbase + j * 16 + colb)] = acc[j][r];
    }
  }
}

// ---------------- host launcher ----------------
extern "C" void kernel_launch(void* const* d_in, const int* in_sizes, int n_in,
                              void* d_out, int out_size, void* d_ws, size_t ws_size,
                              hipStream_t stream) {
  const float* x      = (const float*)d_in[0];
  const float* gate_w = (const float*)d_in[1];
  const int*   Wq_g   = (const int*)d_in[2];
  const float* sc_g   = (const float*)d_in[3];
  const float* zp_g   = (const float*)d_in[4];
  const int*   Wq_u   = (const int*)d_in[5];
  const float* sc_u   = (const float*)d_in[6];
  const float* zp_u   = (const float*)d_in[7];
  const int*   Wq_d   = (const int*)d_in[8];
  const float* sc_d   = (const float*)d_in[9];
  const float* zp_d   = (const float*)d_in[10];
  const float* Wg_sh  = (const float*)d_in[11];
  const float* Wu_sh  = (const float*)d_in[12];
  const float* Wd_sh  = (const float*)d_in[13];
  float* out = (float*)d_out;

  // workspace carve-up
  char* p = (char*)d_ws;
  bf16* xb = (bf16*)p;            p += (size_t)cT * cH * sizeof(bf16);
  int* counts = (int*)p;          p += 256;
  int* offsets = (int*)p;         p += 256;
  int* tid_list = (int*)p;        p += (size_t)cE * cT * sizeof(int);
  float* w_list = (float*)p;      p += (size_t)cE * cT * sizeof(float);
  // S (shared intermediate, 23MB) and Y (routed intermediate, 69MB) are
  // stream-ordered disjoint in time -> share the same region.
  bf16* Sbuf = (bf16*)p;
  bf16* Ybuf = (bf16*)p;

  zero_counts_kernel<<<1, 32, 0, stream>>>(counts);
  cvt_x_kernel<<<(cT * cH + 255) / 256, 256, 0, stream>>>(x, xb, cT * cH);
  gate_kernel<<<cT / 4, 128, 0, stream>>>(x, gate_w, counts, tid_list, w_list);
  prefix_counts_kernel<<<1, 32, 0, stream>>>(counts, offsets);

  shared_mlp1_kernel<<<dim3(cT / TILE_M, cSH / TILE_N), NTHREADS, 0, stream>>>(
      xb, Wg_sh, Wu_sh, Sbuf);
  shared_down_kernel<<<dim3(cT / TILE_M, cH / TILE_N), NTHREADS, 0, stream>>>(
      Sbuf, Wd_sh, out);

  moe_gateup_kernel<<<dim3(cT / TILE_M, cI / TILE_N, cE), NTHREADS, 0, stream>>>(
      xb, Wq_g, sc_g, zp_g, Wq_u, sc_u, zp_u,
      counts, offsets, tid_list, w_list, Ybuf);
  moe_down_kernel<<<dim3(cT / TILE_M, cH / TILE_N, cE), NTHREADS, 0, stream>>>(
      Ybuf, Wq_d, sc_d, zp_d, counts, offsets, tid_list, out);
}
